// MultiHeadAttention_21002390077934
// MI455X (gfx1250) — matrix operand order repaired and checked by reading, not compiled, not run on hardware
//
#include <hip/hip_runtime.h>

#define BB 4
#define TT 2048
#define CC 1024
#define HH 16
#define DD 64
#define BT (BB * TT)   // 8192
#define HD (HH * DD)   // 1024

typedef __attribute__((ext_vector_type(16))) _Float16 v16h;
typedef __attribute__((ext_vector_type(8)))  _Float16 v8h;
typedef __attribute__((ext_vector_type(4)))  _Float16 v4h;
typedef __attribute__((ext_vector_type(8)))  float    v8f;

static __device__ __forceinline__ v8f wmma16(v16h a, v16h b, v8f c) {
  // D = A(16x32 f16) * B(32x16 f16) + C(16x16 f32)
  return __builtin_amdgcn_wmma_f32_16x16x32_f16(
      /*neg_a=*/false, a, /*neg_b=*/false, b,
      /*c_mod=*/(short)0, c, /*reuse_a=*/false, /*reuse_b=*/false);
}

static __device__ __forceinline__ v16h pack16(v8h lo, v8h hi) {
  v16h a;
#pragma unroll
  for (int i = 0; i < 8; ++i) { a[i] = lo[i]; a[8 + i] = hi[i]; }
  return a;
}

static __device__ __forceinline__ v16h ld16(const _Float16* p) {
  return pack16(*(const v8h*)p, *(const v8h*)(p + 16));
}

static __device__ __forceinline__ v8f v8f_zero() {
  v8f z;
#pragma unroll
  for (int i = 0; i < 8; ++i) z[i] = 0.0f;
  return z;
}

// Async global->LDS 16B copy (ASYNCcnt-tracked). Low 32 bits of a generic
// pointer to __shared__ are the LDS byte address (aperture truncation rule).
static __device__ __forceinline__ void async_copy16(void* lds_dst,
                                                    const void* gsrc) {
  unsigned ldsa = (unsigned)(size_t)lds_dst;
  asm volatile("global_load_async_to_lds_b128 %0, %1, off"
               :: "v"(ldsa), "v"(gsrc) : "memory");
}

static __device__ __forceinline__ void wait_async0() {
  asm volatile("s_wait_asynccnt 0x0" ::: "memory");
}

// ---------------------------------------------------------------------------
// fp32 -> fp16 conversion (element count multiple of 4)
// ---------------------------------------------------------------------------
__global__ void cvt_kernel(const float* __restrict__ in,
                           _Float16* __restrict__ out, int n) {
  int i4 = (blockIdx.x * blockDim.x + threadIdx.x) * 4;
  if (i4 + 3 < n) {
    float4 v = *(const float4*)(in + i4);
    v4h o;
    o[0] = (_Float16)v.x; o[1] = (_Float16)v.y;
    o[2] = (_Float16)v.z; o[3] = (_Float16)v.w;
    *(v4h*)(out + i4) = o;
  }
}

// fp32 [H][C][D] -> f16 [H][D][C]   (per-head weight transpose)
__global__ void cvt_t_hcd(const float* __restrict__ in,
                          _Float16* __restrict__ out) {
  int i = blockIdx.x * blockDim.x + threadIdx.x;  // over H*C*D
  int d = i & (DD - 1);
  int c = (i >> 6) & (CC - 1);
  int h = i >> 16;
  out[((size_t)h * DD + d) * CC + c] = (_Float16)in[i];
}

// fp32 [HD][CC] -> f16 [CC][HD]   (Wp transpose)
__global__ void cvt_t_2d(const float* __restrict__ in,
                         _Float16* __restrict__ out) {
  int i = blockIdx.x * blockDim.x + threadIdx.x;  // over HD*CC
  int c = i & (CC - 1);
  int r = i >> 10;
  out[(size_t)c * HD + r] = (_Float16)in[i];
}

// ---------------------------------------------------------------------------
// QKV projection: per (b,h): X[T,C] @ W[C,D] for q,k,v.
// Weights pre-transposed to [H,D,C] so B-operands are contiguous 16B loads.
// V is written transposed [BH, D, T] for the P*V stage of attention.
// Block = 8 waves; wave owns 16 rows; grid.x = 64 * 16 = 1024.
// ---------------------------------------------------------------------------
__global__ __launch_bounds__(256) void qkv_kernel(
    const _Float16* __restrict__ xh,    // [B,T,C]
    const _Float16* __restrict__ wqt,   // [H,D,C]
    const _Float16* __restrict__ wkt,
    const _Float16* __restrict__ wvt,
    _Float16* __restrict__ qh,          // [BH,T,D]
    _Float16* __restrict__ kh,          // [BH,T,D]
    _Float16* __restrict__ vt) {        // [BH,D,T]
  const int bh = blockIdx.x & 63;       // b*16 + h
  const int tchunk = blockIdx.x >> 6;   // 0..15
  const int h = bh & 15;
  const int b = bh >> 4;
  const int wave = threadIdx.x >> 5;
  const int lane = threadIdx.x & 31;
  const int ln = lane & 15;
  const int kg = (lane >> 4) << 3;      // A-layout K sub-offset (0 or 8)
  const int kb = (lane >> 4) << 3;      // B-layout K sub-offset (0 or 8)
  const int row0 = tchunk * 128 + wave * 16;

  const _Float16* xp  = xh + ((size_t)b * TT + row0 + ln) * CC + kg;
  const _Float16* wqp = wqt + (size_t)h * DD * CC;
  const _Float16* wkp = wkt + (size_t)h * DD * CC;
  const _Float16* wvp = wvt + (size_t)h * DD * CC;

  v8f aq[4], ak[4], av[4];
#pragma unroll
  for (int j = 0; j < 4; ++j) { aq[j] = v8f_zero(); ak[j] = v8f_zero(); av[j] = v8f_zero(); }

  for (int kk = 0; kk < CC; kk += 32) {
    // A tile: lane = row; two contiguous 16B chunks (ISA A-layout)
    v16h a = pack16(*(const v8h*)(xp + kk), *(const v8h*)(xp + kk + 16));
#pragma unroll
    for (int j = 0; j < 4; ++j) {
      const size_t brow = (size_t)(j * 16 + ln) * CC + kk + kb;
      aq[j] = wmma16(a, ld16(wqp + brow), aq[j]);
      ak[j] = wmma16(a, ld16(wkp + brow), ak[j]);
      av[j] = wmma16(a, ld16(wvp + brow), av[j]);
    }
  }

  // C/D layout: element r -> row M = r + (lane>=16 ? 8 : 0), col = lane&15
  const int rbase = kg;
  const size_t qbase = ((size_t)bh * TT + row0) * DD;
  const size_t vbase = (size_t)bh * DD * TT;
#pragma unroll
  for (int j = 0; j < 4; ++j)
#pragma unroll
    for (int r = 0; r < 8; ++r) {
      const int rowM = r + rbase;
      const size_t off = qbase + (size_t)rowM * DD + j * 16 + ln;
      qh[off] = (_Float16)aq[j][r];
      kh[off] = (_Float16)ak[j][r];
      vt[vbase + (size_t)(j * 16 + ln) * TT + row0 + rowM] = (_Float16)av[j][r];
    }
}

// ---------------------------------------------------------------------------
// Flash attention: per wave one 16-row Q tile; block streams 32-key K/V tiles
// through double-buffered LDS filled by async global->LDS copies.
// grid.x = 64 * 16 = 1024, block = 256 (8 waves).
// ---------------------------------------------------------------------------
__global__ __launch_bounds__(256) void attn_kernel(
    const _Float16* __restrict__ qh,   // [BH,T,D]
    const _Float16* __restrict__ kh,   // [BH,T,D]
    const _Float16* __restrict__ vt,   // [BH,D,T]
    _Float16* __restrict__ oh) {       // [B,T,HD]
  __shared__ __align__(16) _Float16 sK[2][32][DD];     // 2 x 4 KB
  __shared__ __align__(16) _Float16 sVt[2][DD][32];    // 2 x 4 KB (V transposed)
  __shared__ __align__(16) _Float16 sP[8][16][32];     // 8 KB (per-wave P transpose)

  const int bh = blockIdx.x & 63;
  const int tchunk = blockIdx.x >> 6;
  const int b = bh >> 4, h = bh & 15;
  const int wave = threadIdx.x >> 5;
  const int lane = threadIdx.x & 31;
  const int ln = lane & 15;
  const int kg = (lane >> 4) << 3;
  const int kb = (lane >> 4) << 3;
  const int rbase = kg;
  const int row0 = tchunk * 128 + wave * 16;
  // fold 1/sqrt(64) and log2(e) into one scale; softmax in exp2 domain
  const float qk_scale = 0.125f * 1.44269504f;

  const _Float16* kp = kh + (size_t)bh * TT * DD;
  const _Float16* vp = vt + (size_t)bh * DD * TT;

  // staging coordinates (per thread: one 16B slice of each tile)
  const int skr = threadIdx.x >> 3;             // 0..31 (key row)
  const int skc = (threadIdx.x & 7) << 3;       // 0..56 (d col)
  const int svr = threadIdx.x >> 2;             // 0..63 (d row)
  const int svc = (threadIdx.x & 3) << 3;       // 0..24 (key col)

  // Q A-tiles for d-chunks [0,32) and [32,64)
  const _Float16* qr = qh + ((size_t)bh * TT + row0 + ln) * DD + kg;
  v16h aq0 = pack16(*(const v8h*)qr,        *(const v8h*)(qr + 16));
  v16h aq1 = pack16(*(const v8h*)(qr + 32), *(const v8h*)(qr + 48));

  float m_i[8], l_i[8];
  v8f oacc[4];
#pragma unroll
  for (int r = 0; r < 8; ++r) { m_i[r] = -3.0e38f; l_i[r] = 0.0f; }
#pragma unroll
  for (int j = 0; j < 4; ++j) oacc[j] = v8f_zero();

  // prologue: async-stage tile 0 into buffer 0
  async_copy16(&sK[0][skr][skc],  kp + (size_t)skr * DD + skc);
  async_copy16(&sVt[0][svr][svc], vp + (size_t)svr * TT + svc);

  const int NIT = TT / 32;
  for (int it = 0; it < NIT; ++it) {
    wait_async0();       // own async stores for tile `it` have landed in LDS
    __syncthreads();     // everyone's slices visible; prior buffer reads done
    const int cur = it & 1;

    if (it + 1 < NIT) {  // prefetch next tile into the other buffer
      const int s1 = (it + 1) * 32;
      async_copy16(&sK[1 - cur][skr][skc],  kp + (size_t)(s1 + skr) * DD + skc);
      async_copy16(&sVt[1 - cur][svr][svc], vp + (size_t)svr * TT + s1 + svc);
    }

    // S[16,32] = Q[16,64] * K^T : B-operand rows of K are contiguous in d
    v8f s[2];
    s[0] = v8f_zero(); s[1] = v8f_zero();
#pragma unroll
    for (int jj = 0; jj < 2; ++jj) {
      const _Float16* kr = &sK[cur][jj * 16 + ln][kb];
      s[jj] = wmma16(aq0, ld16(kr),      s[jj]);  // d 0..31
      s[jj] = wmma16(aq1, ld16(kr + 32), s[jj]);  // d 32..63
    }

    // online softmax (exp2 domain): rows live across 16-lane halves
#pragma unroll
    for (int r2 = 0; r2 < 8; ++r2) {
      float x0 = s[0][r2] * qk_scale;
      float x1 = s[1][r2] * qk_scale;
      float cur_m = fmaxf(x0, x1);
#pragma unroll
      for (int mk = 1; mk < 16; mk <<= 1)
        cur_m = fmaxf(cur_m, __shfl_xor(cur_m, mk, 32));
      float mnew = fmaxf(m_i[r2], cur_m);
      float p0 = exp2f(x0 - mnew);
      float p1 = exp2f(x1 - mnew);
      float alpha = exp2f(m_i[r2] - mnew);
      float rs = p0 + p1;
#pragma unroll
      for (int mk = 1; mk < 16; mk <<= 1)
        rs += __shfl_xor(rs, mk, 32);
      l_i[r2] = l_i[r2] * alpha + rs;
      m_i[r2] = mnew;
#pragma unroll
      for (int j = 0; j < 4; ++j) oacc[j][r2] *= alpha;
      // cross-lane transpose of P via per-wave LDS scratch (C/D -> A layout)
      sP[wave][r2 + rbase][ln]      = (_Float16)p0;
      sP[wave][r2 + rbase][16 + ln] = (_Float16)p1;
    }

    // P as A-tile (wave-private LDS; DS ops in-order within a wave)
    const _Float16* pr = &sP[wave][ln][kg];
    v16h ap = pack16(*(const v8h*)pr, *(const v8h*)(pr + 16));

    // O += P[16,32] * V[32,64]; B-operand columns contiguous in sVt
#pragma unroll
    for (int j = 0; j < 4; ++j)
      oacc[j] = wmma16(ap, ld16(&sVt[cur][j * 16 + ln][kb]), oacc[j]);
  }

  // normalize and write O to [B,T,HD]
  const size_t obase = ((size_t)b * TT + row0) * HD + (size_t)h * DD;
#pragma unroll
  for (int r2 = 0; r2 < 8; ++r2) {
    float inv = 1.0f / l_i[r2];
#pragma unroll
    for (int j = 0; j < 4; ++j)
      oh[obase + (size_t)(r2 + rbase) * HD + j * 16 + ln] =
          (_Float16)(oacc[j][r2] * inv);
  }
}

// ---------------------------------------------------------------------------
// Output projection: out[BT,C] = O[BT,HD] @ Wp[HD,C] + bp, fp32 out.
// Wp pre-transposed to [C,HD] so B-operands are contiguous 16B loads.
// grid.x = 64 row-chunks * 16 col-chunks = 1024, block = 256.
// ---------------------------------------------------------------------------
__global__ __launch_bounds__(256) void proj_kernel(
    const _Float16* __restrict__ oh,    // [BT, HD]
    const _Float16* __restrict__ wpt,   // [C, HD]
    const float* __restrict__ bp,       // [C]
    float* __restrict__ out) {          // [BT, C]
  const int rowc = blockIdx.x & 63;
  const int colc = blockIdx.x >> 6;
  const int wave = threadIdx.x >> 5;
  const int lane = threadIdx.x & 31;
  const int ln = lane & 15;
  const int kg = (lane >> 4) << 3;
  const int kb = (lane >> 4) << 3;
  const int row0 = rowc * 128 + wave * 16;
  const int n0b = colc * 64;

  v8f acc[4];
#pragma unroll
  for (int j = 0; j < 4; ++j) acc[j] = v8f_zero();

  const _Float16* op = oh + (size_t)(row0 + ln) * HD + kg;
  for (int kk = 0; kk < HD; kk += 32) {
    v16h a = pack16(*(const v8h*)(op + kk), *(const v8h*)(op + kk + 16));
#pragma unroll
    for (int j = 0; j < 4; ++j) {
      const _Float16* bw = wpt + (size_t)(n0b + j * 16 + ln) * HD + kk + kb;
      acc[j] = wmma16(a, ld16(bw), acc[j]);
    }
  }

  const int rbase = kg;
#pragma unroll
  for (int j = 0; j < 4; ++j) {
    float bias = bp[n0b + j * 16 + ln];
#pragma unroll
    for (int r2 = 0; r2 < 8; ++r2)
      out[(size_t)(row0 + r2 + rbase) * CC + n0b + j * 16 + ln] =
          acc[j][r2] + bias;
  }
}

// ---------------------------------------------------------------------------
extern "C" void kernel_launch(void* const* d_in, const int* in_sizes, int n_in,
                              void* d_out, int out_size, void* d_ws, size_t ws_size,
                              hipStream_t stream) {
  (void)in_sizes; (void)n_in; (void)out_size; (void)ws_size;
  const float* x  = (const float*)d_in[0];
  const float* Wq = (const float*)d_in[1];
  const float* Wk = (const float*)d_in[2];
  const float* Wv = (const float*)d_in[3];
  const float* Wp = (const float*)d_in[4];
  const float* bp = (const float*)d_in[5];
  float* out = (float*)d_out;

  // workspace layout (f16 halves)
  _Float16* w = (_Float16*)d_ws;
  _Float16* xh  = w;                                   // B*T*C   = 8M
  _Float16* wqt = xh  + (size_t)BB * TT * CC;          // [H,D,C] = 1M
  _Float16* wkt = wqt + (size_t)HH * CC * DD;
  _Float16* wvt = wkt + (size_t)HH * CC * DD;
  _Float16* wpt = wvt + (size_t)HH * CC * DD;          // [C,HD]  = 1M
  _Float16* qh  = wpt + (size_t)HD * CC;               // [BH,T,D]= 8M
  _Float16* khb = qh  + (size_t)BB * HH * TT * DD;
  _Float16* vtb = khb + (size_t)BB * HH * TT * DD;     // [BH,D,T]= 8M
  _Float16* ohb = vtb + (size_t)BB * HH * TT * DD;     // [B,T,HD]= 8M

  const int nx = BB * TT * CC;      // 8,388,608
  const int nw = HH * CC * DD;      // 1,048,576
  cvt_kernel<<<nx / 1024, 256, 0, stream>>>(x, xh, nx);
  cvt_t_hcd<<<nw / 256, 256, 0, stream>>>(Wq, wqt);
  cvt_t_hcd<<<nw / 256, 256, 0, stream>>>(Wk, wkt);
  cvt_t_hcd<<<nw / 256, 256, 0, stream>>>(Wv, wvt);
  cvt_t_2d<<<nw / 256, 256, 0, stream>>>(Wp, wpt);

  qkv_kernel<<<1024, 256, 0, stream>>>(xh, wqt, wkt, wvt, qh, khb, vtb);
  attn_kernel<<<1024, 256, 0, stream>>>(qh, khb, vtb, ohb);
  proj_kernel<<<1024, 256, 0, stream>>>(ohb, wpt, bp, out);
}
